// Encoder_81595788689580
// MI455X (gfx1250) — matrix operand order repaired and checked by reading, not compiled, not run on hardware
//
#include <hip/hip_runtime.h>

typedef __bf16 bf16;
typedef __attribute__((ext_vector_type(16))) __bf16 v16bf;
typedef __attribute__((ext_vector_type(8)))  float  v8f;

#define B_  4096
#define T_  24
#define I_  320
#define H_  256
#define H3_ 768
#define MR_ 32              // batch rows per workgroup (2 m-tiles of 16)

// bf16 weight workspace layout (element offsets)
#define OFF_A1  0                       // attn1_W [I, H]
#define OFF_A2  (OFF_A1 + I_ * H_)      // attn2_W [I, I]
#define OFF_A3  (OFF_A2 + I_ * I_)      // attn3_W [I, I]
#define OFF_WIH (OFF_A3 + I_ * I_)      // W_ih    [3H, I]
#define OFF_WHH (OFF_WIH + H3_ * I_)    // W_hh    [3H, H]

__global__ void cvt_f32_bf16(const float* __restrict__ s, bf16* __restrict__ d, int n) {
  int i = blockIdx.x * 256 + threadIdx.x;
  if (i < n) d[i] = (bf16)s[i];
}

__device__ __forceinline__ v16bf frag2(const bf16* p0, const bf16* p1) {
  union { uint4 u[2]; v16bf v; } c;
  c.u[0] = *(const uint4*)p0;   // 8 bf16
  c.u[1] = *(const uint4*)p1;   // 8 bf16
  return c.v;
}

// A-matrix 16x32 bf16 fragment (ISA 7.12.2): lane m=lane&15 holds row m;
// lanes 0-15 K={k0..k0+7, k0+16..k0+23}, lanes 16-31 K={k0+8..k0+15, k0+24..k0+31}.
__device__ __forceinline__ v16bf load_a(const bf16* base, int ld, int m, int k0, int hl) {
  const bf16* p = base + m * ld + k0 + hl * 8;
  return frag2(p, p + 16);
}

// B-matrix 32x16 fragment: column n of B == row n of W[N,K] (X @ W^T).
// Lane n=lane&15 holds K={k0+8h..k0+8h+7, k0+16+8h..k0+16+8h+7}, h=lane>>4 (same as A).
__device__ __forceinline__ v16bf load_b(const bf16* wrow, int k0, int hl) {
  const bf16* p = wrow + k0 + hl * 8;
  return frag2(p, p + 16);
}

#define WMMA(a, b, c) \
  __builtin_amdgcn_wmma_f32_16x16x32_bf16(false, (a), false, (b), (short)0, (c), false, false)

// branch-free transcendentals: v_exp_f32 + v_rcp_f32, saturate correctly at +/-inf
__device__ __forceinline__ float rcp_fast(float v)  { return __builtin_amdgcn_rcpf(v); }
__device__ __forceinline__ float tanh_fast(float v) { return 1.0f - 2.0f * rcp_fast(__expf(2.0f * v) + 1.0f); }
__device__ __forceinline__ float sigm(float v)      { return rcp_fast(1.0f + __expf(-v)); }

__global__ __launch_bounds__(128) void gru_attn_kernel(
    const float* __restrict__ x,  const float* __restrict__ h0,
    const float* __restrict__ b1, const float* __restrict__ b2, const float* __restrict__ b3,
    const float* __restrict__ bih, const float* __restrict__ bhh,
    const bf16* __restrict__ ws,  float* __restrict__ out)
{
  __shared__ __attribute__((aligned(16))) bf16 h_b[2][MR_ * H_];   // ping-pong recurrent state
  __shared__ __attribute__((aligned(16))) bf16 x_b[MR_ * I_];
  __shared__ __attribute__((aligned(16))) float vbuf[MR_ * I_];    // v -> exp(v)
  __shared__ __attribute__((aligned(16))) bf16 t_b[MR_ * I_];      // tanh(w1+w2)
  __shared__ __attribute__((aligned(16))) bf16 wx_b[MR_ * I_];     // x_t * alpha
  __shared__ float red[256];

  const int tid  = threadIdx.x;
  const int lane = tid & 31;
  const int wave = tid >> 5;
  const int hl   = lane >> 4;
  const int nn   = lane & 15;
  const int m0   = blockIdx.x * MR_;

  const bf16* A1  = ws + OFF_A1;
  const bf16* A2  = ws + OFF_A2;
  const bf16* A3  = ws + OFF_A3;
  const bf16* WIH = ws + OFF_WIH;
  const bf16* WHH = ws + OFF_WHH;

  // ---- init: bf16 h in LDS (buffer 0) + f32 h in persistent VGPRs ----
  for (int i = tid; i < MR_ * H_; i += 128) {
    int r = i >> 8, c = i & 255;
    h_b[0][i] = (bf16)h0[(size_t)(m0 + r) * H_ + c];
  }
  float hreg[4][2][8];   // [gate-col tile][m-tile][row]; lane owns (row = r+8*hl, col = g*16+nn)
  #pragma unroll
  for (int gg = 0; gg < 4; ++gg) {
    int c = (wave * 4 + gg) * 16 + nn;
    #pragma unroll
    for (int mt = 0; mt < 2; ++mt)
      #pragma unroll
      for (int r = 0; r < 8; ++r)
        hreg[gg][mt][r] = h0[(size_t)(m0 + mt * 16 + r + 8 * hl) * H_ + c];
  }

  int cur = 0;
  #pragma unroll 1
  for (int t = 0; t < T_; ++t) {
    const bf16* hb  = h_b[cur];
    bf16*       hbn = h_b[cur ^ 1];
    __syncthreads();

    // ---- stage x_t tile as bf16 (float4 global loads, 8B LDS stores) ----
    for (int i4 = tid; i4 < MR_ * (I_ / 4); i4 += 128) {
      int r  = i4 / (I_ / 4);
      int c4 = (i4 - r * (I_ / 4)) * 4;
      float4 f = *(const float4*)(x + ((size_t)(m0 + r) * T_ + t) * I_ + c4);
      bf16 t4[4] = {(bf16)f.x, (bf16)f.y, (bf16)f.z, (bf16)f.w};
      *(uint2*)(x_b + r * I_ + c4) = *(const uint2*)t4;
    }
    __syncthreads();

    // ---- GEMM A: w12 = h @ A1^T + x_t @ A2^T + b1 + b2; tanh fused in epilogue ----
    #pragma unroll 1
    for (int j = 0; j < 5; ++j) {
      int n = (wave * 5 + j) * 16 + nn;
      const bf16* w1r = A1 + n * H_;
      const bf16* w2r = A2 + n * I_;
      float bias = b1[n] + b2[n];
      v8f a0, a1;
      #pragma unroll
      for (int r = 0; r < 8; ++r) { a0[r] = bias; a1[r] = bias; }
      #pragma unroll 2
      for (int k0 = 0; k0 < H_; k0 += 32) {
        v16bf bb = load_b(w1r, k0, hl);
        a0 = WMMA(load_a(hb, H_, nn, k0, hl), bb, a0);
        a1 = WMMA(load_a(hb, H_, 16 + nn, k0, hl), bb, a1);
      }
      #pragma unroll 2
      for (int k0 = 0; k0 < I_; k0 += 32) {
        v16bf bb = load_b(w2r, k0, hl);
        a0 = WMMA(load_a(x_b, I_, nn, k0, hl), bb, a0);
        a1 = WMMA(load_a(x_b, I_, 16 + nn, k0, hl), bb, a1);
      }
      #pragma unroll
      for (int r = 0; r < 8; ++r) {
        t_b[(r + 8 * hl) * I_ + n]      = (bf16)tanh_fast(a0[r]);
        t_b[(16 + r + 8 * hl) * I_ + n] = (bf16)tanh_fast(a1[r]);
      }
    }
    __syncthreads();

    // ---- GEMM B: v = tanh(w12) @ A3^T + b3 -> vbuf ----
    #pragma unroll 1
    for (int j = 0; j < 5; ++j) {
      int n = (wave * 5 + j) * 16 + nn;
      const bf16* w3r = A3 + n * I_;
      float bias = b3[n];
      v8f a0, a1;
      #pragma unroll
      for (int r = 0; r < 8; ++r) { a0[r] = bias; a1[r] = bias; }
      #pragma unroll 2
      for (int k0 = 0; k0 < I_; k0 += 32) {
        v16bf bb = load_b(w3r, k0, hl);
        a0 = WMMA(load_a(t_b, I_, nn, k0, hl), bb, a0);
        a1 = WMMA(load_a(t_b, I_, 16 + nn, k0, hl), bb, a1);
      }
      #pragma unroll
      for (int r = 0; r < 8; ++r) {
        vbuf[(r + 8 * hl) * I_ + n]      = a0[r];
        vbuf[(16 + r + 8 * hl) * I_ + n] = a1[r];
      }
    }
    __syncthreads();

    // ---- softmax over I per row (4 threads/row, contiguous 80-elem chunks); wx = x_t*alpha ----
    {
      int row = tid >> 2, sub = tid & 3;
      float* vr = vbuf + row * I_ + sub * 80;
      float mx = -3.4e38f;
      for (int k = 0; k < 20; ++k) {
        float4 f = *(const float4*)(vr + 4 * k);
        mx = fmaxf(fmaxf(fmaxf(mx, f.x), fmaxf(f.y, f.z)), f.w);
      }
      red[tid] = mx;
      __syncthreads();
      float rm = red[row * 4];
      #pragma unroll
      for (int q = 1; q < 4; ++q) rm = fmaxf(rm, red[row * 4 + q]);
      float s = 0.f;
      for (int k = 0; k < 20; ++k) {
        float4 f = *(const float4*)(vr + 4 * k);
        f.x = __expf(f.x - rm); f.y = __expf(f.y - rm);
        f.z = __expf(f.z - rm); f.w = __expf(f.w - rm);
        s += (f.x + f.y) + (f.z + f.w);
        *(float4*)(vr + 4 * k) = f;
      }
      red[128 + tid] = s;
      __syncthreads();
      float rs = 0.f;
      #pragma unroll
      for (int q = 0; q < 4; ++q) rs += red[128 + row * 4 + q];
      float inv = rcp_fast(rs);
      const bf16* xr = x_b + row * I_ + sub * 80;
      bf16*       wr = wx_b + row * I_ + sub * 80;
      for (int k = 0; k < 20; ++k) {
        float4 f = *(const float4*)(vr + 4 * k);
        bf16 x4[4];
        *(uint2*)x4 = *(const uint2*)(xr + 4 * k);
        bf16 o4[4] = { (bf16)((float)x4[0] * (f.x * inv)), (bf16)((float)x4[1] * (f.y * inv)),
                       (bf16)((float)x4[2] * (f.z * inv)), (bf16)((float)x4[3] * (f.w * inv)) };
        *(uint2*)(wr + 4 * k) = *(const uint2*)o4;
      }
    }
    __syncthreads();

    // ---- gate GEMMs + GRU update, all in registers ----
    // wave owns gate columns c = (wave*4+gg)*16 + nn for gg=0..3; r/z share one
    // accumulator (gi+gh with summed bias), n-gate keeps i_n and h_n separate.
    #pragma unroll
    for (int gg = 0; gg < 4; ++gg) {
      int c0  = (wave * 4 + gg) * 16;
      int n_r = c0 + nn, n_z = 256 + c0 + nn, n_n = 512 + c0 + nn;
      const bf16 *ir = WIH + (size_t)n_r * I_, *iz = WIH + (size_t)n_z * I_, *in_ = WIH + (size_t)n_n * I_;
      const bf16 *hr = WHH + (size_t)n_r * H_, *hz = WHH + (size_t)n_z * H_, *hn_ = WHH + (size_t)n_n * H_;
      float br = bih[n_r] + bhh[n_r];
      float bz = bih[n_z] + bhh[n_z];
      float bn = bih[n_n];
      float bm = bhh[n_n];
      v8f ar0, ar1, az0, az1, an0, an1, am0, am1;
      #pragma unroll
      for (int r = 0; r < 8; ++r) {
        ar0[r] = br; ar1[r] = br; az0[r] = bz; az1[r] = bz;
        an0[r] = bn; an1[r] = bn; am0[r] = bm; am1[r] = bm;
      }
      #pragma unroll 2
      for (int k0 = 0; k0 < I_; k0 += 32) {     // gi part: wx @ W_ih^T
        v16bf Aa = load_a(wx_b, I_, nn, k0, hl);
        v16bf Ab = load_a(wx_b, I_, 16 + nn, k0, hl);
        v16bf Br = load_b(ir, k0, hl);
        v16bf Bz = load_b(iz, k0, hl);
        v16bf Bn = load_b(in_, k0, hl);
        ar0 = WMMA(Aa, Br, ar0); ar1 = WMMA(Ab, Br, ar1);
        az0 = WMMA(Aa, Bz, az0); az1 = WMMA(Ab, Bz, az1);
        an0 = WMMA(Aa, Bn, an0); an1 = WMMA(Ab, Bn, an1);
      }
      #pragma unroll 2
      for (int k0 = 0; k0 < H_; k0 += 32) {     // gh part: h @ W_hh^T
        v16bf Aa = load_a(hb, H_, nn, k0, hl);
        v16bf Ab = load_a(hb, H_, 16 + nn, k0, hl);
        v16bf Br = load_b(hr, k0, hl);
        v16bf Bz = load_b(hz, k0, hl);
        v16bf Bn = load_b(hn_, k0, hl);
        ar0 = WMMA(Aa, Br, ar0); ar1 = WMMA(Ab, Br, ar1);
        az0 = WMMA(Aa, Bz, az0); az1 = WMMA(Ab, Bz, az1);
        am0 = WMMA(Aa, Bn, am0); am1 = WMMA(Ab, Bn, am1);
      }
      int c = c0 + nn;
      #pragma unroll
      for (int r = 0; r < 8; ++r) {
        int row0 = r + 8 * hl;
        {
          float rg = sigm(ar0[r]), zg = sigm(az0[r]);
          float ng = tanh_fast(an0[r] + rg * am0[r]);
          float hv = (1.0f - zg) * ng + zg * hreg[gg][0][r];
          hreg[gg][0][r] = hv;
          hbn[row0 * H_ + c] = (bf16)hv;
          out[((size_t)(m0 + row0) * T_ + t) * H_ + c] = hv;
        }
        {
          float rg = sigm(ar1[r]), zg = sigm(az1[r]);
          float ng = tanh_fast(an1[r] + rg * am1[r]);
          float hv = (1.0f - zg) * ng + zg * hreg[gg][1][r];
          hreg[gg][1][r] = hv;
          hbn[(16 + row0) * H_ + c] = (bf16)hv;
          out[((size_t)(m0 + 16 + row0) * T_ + t) * H_ + c] = hv;
        }
      }
    }
    cur ^= 1;
  }
}

extern "C" void kernel_launch(void* const* d_in, const int* in_sizes, int n_in,
                              void* d_out, int out_size, void* d_ws, size_t ws_size,
                              hipStream_t stream) {
  const float* x   = (const float*)d_in[0];
  const float* h0  = (const float*)d_in[1];
  const float* a1W = (const float*)d_in[2];
  const float* a1b = (const float*)d_in[3];
  const float* a2W = (const float*)d_in[4];
  const float* a2b = (const float*)d_in[5];
  const float* a3W = (const float*)d_in[6];
  const float* a3b = (const float*)d_in[7];
  const float* Wih = (const float*)d_in[8];
  const float* bih = (const float*)d_in[9];
  const float* Whh = (const float*)d_in[10];
  const float* bhh = (const float*)d_in[11];
  float* out = (float*)d_out;
  bf16* ws   = (bf16*)d_ws;

  cvt_f32_bf16<<<(I_ * H_ + 255) / 256, 256, 0, stream>>>(a1W, ws + OFF_A1, I_ * H_);
  cvt_f32_bf16<<<(I_ * I_ + 255) / 256, 256, 0, stream>>>(a2W, ws + OFF_A2, I_ * I_);
  cvt_f32_bf16<<<(I_ * I_ + 255) / 256, 256, 0, stream>>>(a3W, ws + OFF_A3, I_ * I_);
  cvt_f32_bf16<<<(H3_ * I_ + 255) / 256, 256, 0, stream>>>(Wih, ws + OFF_WIH, H3_ * I_);
  cvt_f32_bf16<<<(H3_ * H_ + 255) / 256, 256, 0, stream>>>(Whh, ws + OFF_WHH, H3_ * H_);

  gru_attn_kernel<<<B_ / MR_, 128, 0, stream>>>(x, h0, a1b, a2b, a3b, bih, bhh, ws, out);
}
